// ChromophoreSolventTransformerGNN_GAP_GMP_10900626997629
// MI455X (gfx1250) — matrix operand order repaired and checked by reading, not compiled, not run on hardware
//
#include <hip/hip_runtime.h>
#include <math.h>

// ---------------- problem constants ----------------
#define NNODES 15000
#define NPAD   15008          // 15000 padded up to multiple of 16
#define NEDGES 150000
#define NGRAPH 64
#define FNODE  64
#define FEDGE  16
#define FSOL   128
#define HEADS  4
#define HC1    512            // H1*C1
#define HC2    1024           // H2*C2
#define EPSBN  1e-5f

typedef __attribute__((ext_vector_type(16))) _Float16 v16h;
typedef __attribute__((ext_vector_type(8)))  float    v8f;

// ---------------- small device helpers ----------------
__device__ __forceinline__ unsigned f2k(float f) {
  unsigned u = __float_as_uint(f);
  return (u & 0x80000000u) ? ~u : (u | 0x80000000u);
}
__device__ __forceinline__ float k2f(unsigned k) {
  unsigned u = (k & 0x80000000u) ? (k & 0x7FFFFFFFu) : ~k;
  return __uint_as_float(u);
}

// ---------------- fills ----------------
__global__ void fill_f32(float* p, float v, int n) {
  int i = blockIdx.x * blockDim.x + threadIdx.x;
  if (i < n) p[i] = v;
}
__global__ void fill_u32(unsigned* p, unsigned v, int n) {
  int i = blockIdx.x * blockDim.x + threadIdx.x;
  if (i < n) p[i] = v;
}

// ---------------- fragment packing (fp32 -> f16, WMMA-native layout) ----------------
// A [Mpad,K] -> tiles of 16x32; tile t = mt*ktiles + kt; within tile, lane L owns
// 16 contiguous halfs matching the v_wmma_f32_16x16x32_f16 A-operand register layout.
__global__ void pack_a_f16(const float* __restrict__ s, _Float16* __restrict__ d,
                           int M, int Mpad, int K) {
  int i = blockIdx.x * blockDim.x + threadIdx.x;
  if (i >= Mpad * K) return;
  int t  = i >> 9;            // /512 elems per tile
  int r  = i & 511;
  int L  = r >> 4, j = r & 15;
  int iv = j >> 1, half = j & 1;
  int m  = L & 15, hs = L >> 4;
  int ka = ((iv < 4) ? (iv * 2) : (16 + (iv - 4) * 2)) + hs * 8 + half;
  int ktiles = K >> 5;
  int mt = t / ktiles, kt = t - mt * ktiles;
  int row = (mt << 4) + m;
  d[i] = (row < M) ? (_Float16)s[(size_t)row * K + (kt << 5) + ka] : (_Float16)0.0f;
}
// B [K,N] -> tiles of 32x16; tile t = kt*tilesN + nt; lane L owns 16 contiguous halfs
// matching the B-operand register layout (lanes 0-15: K=0..7,16..23, lanes 16-31: K=8..15,24..31).
__global__ void pack_b_f16(const float* __restrict__ s, _Float16* __restrict__ d,
                           int K, int N) {
  int i = blockIdx.x * blockDim.x + threadIdx.x;
  if (i >= K * N) return;
  int t  = i >> 9;
  int r  = i & 511;
  int L  = r >> 4, j = r & 15;
  int iv = j >> 1, half = j & 1;
  int n  = L & 15, hs = L >> 4;
  int kb = ((iv < 4) ? (iv * 2) : (16 + (iv - 4) * 2)) + hs * 8 + half;
  int tilesN = N >> 4;
  int kt = t / tilesN, nt = t - kt * tilesN;
  d[i] = (_Float16)s[(size_t)((kt << 5) + kb) * N + (nt << 4) + n];
}

// ---------------- WMMA GEMM: C[M,N] = A*B + bias ----------------
// one wave per 16(M) x 64(N) output block: 4 N-tiles share one A fragment.
// Fragments are pre-packed: per tile, lane L loads 32 contiguous bytes (2x b128).
__global__ void wmma_gemm_bias(const _Float16* __restrict__ Ap, const _Float16* __restrict__ Bp,
                               const float* __restrict__ bias, float* __restrict__ C,
                               int Mpad, int Mvalid, int N, int K, int ldc) {
  int wid  = (blockIdx.x * blockDim.x + threadIdx.x) >> 5;
  int lane = threadIdx.x & 31;
  int ktiles = K >> 5;
  int tilesN = N >> 4;
  int ngrp   = N >> 6;               // groups of 4 n-tiles
  int mt = wid / ngrp;
  int ng = wid - mt * ngrp;
  if (mt >= (Mpad >> 4)) return;
  int nt0 = ng << 2;

  const v16h* apv = (const v16h*)Ap;   // 32 v16h per tile
  const v16h* bpv = (const v16h*)Bp;

  v8f c0 = {}, c1 = {}, c2 = {}, c3 = {};
  size_t abase = ((size_t)mt * ktiles) * 32 + lane;

#pragma unroll 2
  for (int kt = 0; kt < ktiles; ++kt) {
    v16h a = apv[abase + (size_t)kt * 32];
    size_t bbase = ((size_t)kt * tilesN + nt0) * 32 + lane;
    v16h b0 = bpv[bbase];
    v16h b1 = bpv[bbase + 32];
    v16h b2 = bpv[bbase + 64];
    v16h b3 = bpv[bbase + 96];
    c0 = __builtin_amdgcn_wmma_f32_16x16x32_f16(false, a, false, b0, (short)0, c0, false, false);
    c1 = __builtin_amdgcn_wmma_f32_16x16x32_f16(false, a, false, b1, (short)0, c1, false, false);
    c2 = __builtin_amdgcn_wmma_f32_16x16x32_f16(false, a, false, b2, (short)0, c2, false, false);
    c3 = __builtin_amdgcn_wmma_f32_16x16x32_f16(false, a, false, b3, (short)0, c3, false, false);
  }

  int hs = lane >> 4, n16 = lane & 15;
  v8f acc[4] = {c0, c1, c2, c3};
#pragma unroll
  for (int j = 0; j < 4; ++j) {
    int col  = ((nt0 + j) << 4) + n16;
    float bv = bias[col];
#pragma unroll
    for (int r = 0; r < 8; ++r) {
      int row = (mt << 4) + r + (hs << 3);
      if (row < Mvalid) C[(size_t)row * ldc + col] = acc[j][r] + bv;
    }
  }
}

// ---------------- edge attention: one wave per (edge, head) ----------------
__global__ void edge_alpha(const float* __restrict__ q, const float* __restrict__ k,
                           const float* __restrict__ eattr, const float* __restrict__ We,
                           const int* __restrict__ src, const int* __restrict__ dst,
                           float* __restrict__ alpha, unsigned* __restrict__ amax,
                           int E, int H, int C, int D, float scale) {
  int wid  = (blockIdx.x * blockDim.x + threadIdx.x) >> 5;
  int lane = threadIdx.x & 31;
  if (wid >= E * H) return;
  int e = wid / H, h = wid - e * H;
  int s = src[e], d = dst[e];
  int hC = h * C;
  const float* ea = eattr + (size_t)e * FEDGE;
  const float* qd = q + (size_t)d * D + hC;
  const float* ks = k + (size_t)s * D + hC;
  float acc = 0.0f;
  for (int c = lane; c < C; c += 32) {
    float ev = 0.0f;
#pragma unroll
    for (int j = 0; j < FEDGE; ++j) ev += ea[j] * We[(size_t)j * D + hC + c];
    acc += qd[c] * (ks[c] + ev);
  }
#pragma unroll
  for (int off = 16; off > 0; off >>= 1) acc += __shfl_xor(acc, off, 32);
  if (lane == 0) {
    float a = acc * scale;
    alpha[(size_t)e * H + h] = a;
    atomicMax(&amax[(size_t)d * H + h], f2k(a));
  }
}

__global__ void edge_expsum(const float* __restrict__ alpha, float* __restrict__ exb,
                            const int* __restrict__ dst, const unsigned* __restrict__ amax,
                            float* __restrict__ den, int E, int H) {
  int i = blockIdx.x * blockDim.x + threadIdx.x;
  if (i >= E * H) return;
  int e = i / H, h = i - e * H;
  int d = dst[e];
  float ex = expf(alpha[i] - k2f(amax[(size_t)d * H + h]));
  exb[i] = ex;
  atomicAdd(&den[(size_t)d * H + h], ex);
}

__global__ void edge_scatter(const float* __restrict__ v, const float* __restrict__ eattr,
                             const float* __restrict__ We, const int* __restrict__ src,
                             const int* __restrict__ dst, const float* __restrict__ exb,
                             const float* __restrict__ den, float* __restrict__ out,
                             int E, int H, int C, int D) {
  int wid  = (blockIdx.x * blockDim.x + threadIdx.x) >> 5;
  int lane = threadIdx.x & 31;
  if (wid >= E * H) return;
  int e = wid / H, h = wid - e * H;
  int s = src[e], d = dst[e];
  int hC = h * C;
  float a = exb[(size_t)e * H + h] / (den[(size_t)d * H + h] + 1e-16f);
  const float* ea = eattr + (size_t)e * FEDGE;
  const float* vs = v + (size_t)s * D + hC;
  float* od = out + (size_t)d * D + hC;
  for (int c = lane; c < C; c += 32) {
    float ev = 0.0f;
#pragma unroll
    for (int j = 0; j < FEDGE; ++j) ev += ea[j] * We[(size_t)j * D + hC + c];
    atomicAdd(&od[c], (vs[c] + ev) * a);
  }
}

// ---------------- batchnorm (batch statistics) ----------------
__global__ void bn_stats(const float* __restrict__ x, float* __restrict__ sum,
                         float* __restrict__ sumsq, int Nrows, int D,
                         int colChunks, int rowsPerChunk) {
  int cc  = blockIdx.x % colChunks;
  int rc  = blockIdx.x / colChunks;
  int tid = threadIdx.x;
  int col = cc * 64 + (tid & 63);
  int rend = (rc + 1) * rowsPerChunk; if (rend > Nrows) rend = Nrows;
  float s = 0.0f, s2 = 0.0f;
  for (int r = rc * rowsPerChunk + (tid >> 6); r < rend; r += 4) {
    float v = x[(size_t)r * D + col];
    s += v; s2 += v * v;
  }
  atomicAdd(&sum[col], s);
  atomicAdd(&sumsq[col], s2);
}
__global__ void bn_apply_relu(float* __restrict__ x, const float* __restrict__ sum,
                              const float* __restrict__ sumsq, const float* __restrict__ g,
                              const float* __restrict__ b, int Nrows, int D) {
  int i = blockIdx.x * blockDim.x + threadIdx.x;
  if (i >= Nrows * D) return;
  int col = i % D;
  float m  = sum[col] / (float)Nrows;
  float vr = sumsq[col] / (float)Nrows - m * m;
  float y  = g[col] * (x[i] - m) * rsqrtf(vr + EPSBN) + b[col];
  x[i] = fmaxf(y, 0.0f);
}

// ---------------- global mean/max pooling ----------------
__global__ void pool_count(const int* __restrict__ batch, float* __restrict__ cnt, int Nrows) {
  int r = blockIdx.x * blockDim.x + threadIdx.x;
  if (r < Nrows) atomicAdd(&cnt[batch[r]], 1.0f);
}
__global__ void pool_scatter(const float* __restrict__ h, const int* __restrict__ batch,
                             float* __restrict__ gap, unsigned* __restrict__ gmp,
                             int Nrows, int D) {
  int i = blockIdx.x * blockDim.x + threadIdx.x;
  if (i >= Nrows * D) return;
  int r = i / D, d = i - r * D;
  int g = batch[r];
  float v = h[i];
  atomicAdd(&gap[(size_t)g * D + d], v);
  atomicMax(&gmp[(size_t)g * D + d], f2k(v));
}
__global__ void pool_finalize(float* __restrict__ gap, unsigned* __restrict__ gmp_u,
                              const float* __restrict__ cnt, int G, int D) {
  int i = blockIdx.x * blockDim.x + threadIdx.x;
  if (i >= G * D) return;
  int g = i / D;
  gap[i] /= fmaxf(cnt[g], 1.0f);
  float gm = k2f(gmp_u[i]);
  if (!(gm > -1e38f && gm < 1e38f)) gm = 0.0f;   // non-finite -> 0
  ((float*)gmp_u)[i] = gm;
}

// ---------------- small dense layers ----------------
__global__ void dense_bias(const float* __restrict__ X, const float* __restrict__ W,
                           const float* __restrict__ b, float* __restrict__ Y,
                           int M, int K, int N, int doRelu) {
  int i = blockIdx.x * blockDim.x + threadIdx.x;
  if (i >= M * N) return;
  int m = i / N, n = i - m * N;
  float acc = b[n];
  for (int k = 0; k < K; ++k) acc += X[(size_t)m * K + k] * W[(size_t)k * N + n];
  Y[i] = doRelu ? fmaxf(acc, 0.0f) : acc;
}
__global__ void fc1_concat(const float* __restrict__ gap, const float* __restrict__ gmp,
                           const float* __restrict__ sol, const float* __restrict__ W,
                           const float* __restrict__ b, float* __restrict__ Y) {
  int i = blockIdx.x * blockDim.x + threadIdx.x;
  if (i >= NGRAPH * 128) return;
  int g = i / 128, j = i - g * 128;
  float acc = b[j];
  for (int k = 0; k < HC2; ++k)  acc += gap[(size_t)g * HC2 + k] * W[(size_t)k * 128 + j];
  for (int k = 0; k < HC2; ++k)  acc += gmp[(size_t)g * HC2 + k] * W[(size_t)(HC2 + k) * 128 + j];
  for (int k = 0; k < FSOL; ++k) acc += sol[(size_t)g * FSOL + k] * W[(size_t)(2 * HC2 + k) * 128 + j];
  Y[i] = acc;
}
__global__ void bn_small_relu(float* __restrict__ z, const float* __restrict__ g,
                              const float* __restrict__ b, int M, int D) {
  int col = threadIdx.x;
  if (col >= D) return;
  float s = 0.0f, s2 = 0.0f;
  for (int r = 0; r < M; ++r) { float v = z[(size_t)r * D + col]; s += v; s2 += v * v; }
  float m  = s / (float)M;
  float vr = s2 / (float)M - m * m;
  float inv = rsqrtf(vr + EPSBN);
  for (int r = 0; r < M; ++r) {
    float y = g[col] * (z[(size_t)r * D + col] - m) * inv + b[col];
    z[(size_t)r * D + col] = fmaxf(y, 0.0f);
  }
}

// ---------------- host side ----------------
static inline void* wsTake(char*& p, size_t bytes) {
  void* r = p;
  p += (bytes + 255) & ~(size_t)255;
  return r;
}
static void run_gemm(const _Float16* Ap, const float* W, const float* bias, float* C,
                     int Mpad, int M, int N, int K, int ldc, _Float16* Bp, hipStream_t st) {
  int nW = K * N;
  pack_b_f16<<<(nW + 255) / 256, 256, 0, st>>>(W, Bp, K, N);
  int waves  = (Mpad / 16) * (N / 64);
  int blocks = (waves + 7) / 8;                 // 8 waves (256 threads) per block
  wmma_gemm_bias<<<blocks, 256, 0, st>>>(Ap, Bp, bias, C, Mpad, M, N, K, ldc);
}

extern "C" void kernel_launch(void* const* d_in, const int* in_sizes, int n_in,
                              void* d_out, int out_size, void* d_ws, size_t ws_size,
                              hipStream_t stream) {
  // ---- inputs (setup_inputs order) ----
  const float* x      = (const float*)d_in[0];
  const float* eattr  = (const float*)d_in[1];
  const float* solfp  = (const float*)d_in[2];
  const int*   eidx   = (const int*)  d_in[3];
  const int*   batch  = (const int*)  d_in[4];
  const float* Wq1 = (const float*)d_in[5];  const float* bq1 = (const float*)d_in[6];
  const float* Wk1 = (const float*)d_in[7];  const float* bk1 = (const float*)d_in[8];
  const float* Wv1 = (const float*)d_in[9];  const float* bv1 = (const float*)d_in[10];
  const float* We1 = (const float*)d_in[11];
  const float* Ws1 = (const float*)d_in[12]; const float* bs1 = (const float*)d_in[13];
  const float* g1  = (const float*)d_in[14]; const float* be1 = (const float*)d_in[15];
  const float* Wq2 = (const float*)d_in[16]; const float* bq2 = (const float*)d_in[17];
  const float* Wk2 = (const float*)d_in[18]; const float* bk2 = (const float*)d_in[19];
  const float* Wv2 = (const float*)d_in[20]; const float* bv2 = (const float*)d_in[21];
  const float* We2 = (const float*)d_in[22];
  const float* Ws2 = (const float*)d_in[23]; const float* bs2 = (const float*)d_in[24];
  const float* g2  = (const float*)d_in[25]; const float* be2 = (const float*)d_in[26];
  const float* Wsol = (const float*)d_in[27]; const float* bsol = (const float*)d_in[28];
  const float* Wfc1 = (const float*)d_in[29]; const float* bfc1 = (const float*)d_in[30];
  const float* gf   = (const float*)d_in[31]; const float* bf   = (const float*)d_in[32];
  const float* Wfc2 = (const float*)d_in[33]; const float* bfc2 = (const float*)d_in[34];
  const int* src = eidx;            // edge_index[0]
  const int* dst = eidx + NEDGES;   // edge_index[1]

  // ---- workspace carve-out (deterministic) ----
  char* p = (char*)d_ws;
  _Float16* Ap    = (_Float16*)wsTake(p, (size_t)NPAD * 512 * 2);   // packed A (max K=512)
  _Float16* Bp    = (_Float16*)wsTake(p, (size_t)512 * 1024 * 2);   // packed weights
  float* qb   = (float*)wsTake(p, (size_t)NPAD * HC2 * 4);
  float* kb   = (float*)wsTake(p, (size_t)NPAD * HC2 * 4);
  float* vb   = (float*)wsTake(p, (size_t)NPAD * HC2 * 4);
  float* hb   = (float*)wsTake(p, (size_t)NPAD * HC2 * 4);
  float* alp  = (float*)wsTake(p, (size_t)NEDGES * HEADS * 4);
  float* exb  = (float*)wsTake(p, (size_t)NEDGES * HEADS * 4);
  unsigned* amax = (unsigned*)wsTake(p, (size_t)NPAD * HEADS * 4);
  float* den  = (float*)wsTake(p, (size_t)NPAD * HEADS * 4);
  float* ssum = (float*)wsTake(p, 1024 * 4);
  float* ssq  = (float*)wsTake(p, 1024 * 4);
  float* gap  = (float*)wsTake(p, (size_t)NGRAPH * HC2 * 4);
  unsigned* gmp = (unsigned*)wsTake(p, (size_t)NGRAPH * HC2 * 4);
  float* cnt  = (float*)wsTake(p, NGRAPH * 4);
  float* sol  = (float*)wsTake(p, (size_t)NGRAPH * 128 * 4);
  float* zfc  = (float*)wsTake(p, (size_t)NGRAPH * 128 * 4);

  const int ewaves  = NEDGES * HEADS;           // 600000 waves
  const int eblocks = ewaves / 8;               // 8 waves/block
  const int ethreads = (ewaves + 255) / 256;

  // ================= layer 1 =================
  {
    int n = NPAD * FNODE;
    pack_a_f16<<<(n + 255) / 256, 256, 0, stream>>>(x, Ap, NNODES, NPAD, FNODE);
  }
  run_gemm(Ap, Wq1, bq1, qb, NPAD, NNODES, HC1, FNODE, HC1, Bp, stream);
  run_gemm(Ap, Wk1, bk1, kb, NPAD, NNODES, HC1, FNODE, HC1, Bp, stream);
  run_gemm(Ap, Wv1, bv1, vb, NPAD, NNODES, HC1, FNODE, HC1, Bp, stream);
  run_gemm(Ap, Ws1, bs1, hb, NPAD, NNODES, HC1, FNODE, HC1, Bp, stream);   // skip into out

  fill_u32<<<(NPAD * HEADS + 255) / 256, 256, 0, stream>>>(amax, 0u, NPAD * HEADS);
  fill_f32<<<(NPAD * HEADS + 255) / 256, 256, 0, stream>>>(den, 0.0f, NPAD * HEADS);
  edge_alpha<<<eblocks, 256, 0, stream>>>(qb, kb, eattr, We1, src, dst, alp, amax,
                                          NEDGES, HEADS, 128, HC1, 1.0f / sqrtf(128.0f));
  edge_expsum<<<ethreads, 256, 0, stream>>>(alp, exb, dst, amax, den, NEDGES, HEADS);
  edge_scatter<<<eblocks, 256, 0, stream>>>(vb, eattr, We1, src, dst, exb, den, hb,
                                            NEDGES, HEADS, 128, HC1);
  // BN1 + relu (in place on hb, [15000,512])
  fill_f32<<<(HC1 + 255) / 256, 256, 0, stream>>>(ssum, 0.0f, HC1);
  fill_f32<<<(HC1 + 255) / 256, 256, 0, stream>>>(ssq, 0.0f, HC1);
  bn_stats<<<(HC1 / 64) * 64, 256, 0, stream>>>(hb, ssum, ssq, NNODES, HC1, HC1 / 64,
                                                (NNODES + 63) / 64);
  bn_apply_relu<<<(NNODES * HC1 + 255) / 256, 256, 0, stream>>>(hb, ssum, ssq, g1, be1,
                                                                NNODES, HC1);

  // ================= layer 2 =================
  {
    int n = NPAD * HC1;
    pack_a_f16<<<(n + 255) / 256, 256, 0, stream>>>(hb, Ap, NNODES, NPAD, HC1);
  }
  run_gemm(Ap, Wq2, bq2, qb, NPAD, NNODES, HC2, HC1, HC2, Bp, stream);
  run_gemm(Ap, Wk2, bk2, kb, NPAD, NNODES, HC2, HC1, HC2, Bp, stream);
  run_gemm(Ap, Wv2, bv2, vb, NPAD, NNODES, HC2, HC1, HC2, Bp, stream);
  run_gemm(Ap, Ws2, bs2, hb, NPAD, NNODES, HC2, HC1, HC2, Bp, stream);     // overwrites h1

  fill_u32<<<(NPAD * HEADS + 255) / 256, 256, 0, stream>>>(amax, 0u, NPAD * HEADS);
  fill_f32<<<(NPAD * HEADS + 255) / 256, 256, 0, stream>>>(den, 0.0f, NPAD * HEADS);
  edge_alpha<<<eblocks, 256, 0, stream>>>(qb, kb, eattr, We2, src, dst, alp, amax,
                                          NEDGES, HEADS, 256, HC2, 1.0f / sqrtf(256.0f));
  edge_expsum<<<ethreads, 256, 0, stream>>>(alp, exb, dst, amax, den, NEDGES, HEADS);
  edge_scatter<<<eblocks, 256, 0, stream>>>(vb, eattr, We2, src, dst, exb, den, hb,
                                            NEDGES, HEADS, 256, HC2);
  // BN2 + relu
  fill_f32<<<(HC2 + 255) / 256, 256, 0, stream>>>(ssum, 0.0f, HC2);
  fill_f32<<<(HC2 + 255) / 256, 256, 0, stream>>>(ssq, 0.0f, HC2);
  bn_stats<<<(HC2 / 64) * 64, 256, 0, stream>>>(hb, ssum, ssq, NNODES, HC2, HC2 / 64,
                                                (NNODES + 63) / 64);
  bn_apply_relu<<<(NNODES * HC2 + 255) / 256, 256, 0, stream>>>(hb, ssum, ssq, g2, be2,
                                                                NNODES, HC2);

  // ================= pooling + head =================
  fill_f32<<<(NGRAPH * HC2 + 255) / 256, 256, 0, stream>>>(gap, 0.0f, NGRAPH * HC2);
  fill_u32<<<(NGRAPH * HC2 + 255) / 256, 256, 0, stream>>>(gmp, 0u, NGRAPH * HC2);
  fill_f32<<<1, 256, 0, stream>>>(cnt, 0.0f, NGRAPH);
  pool_count<<<(NNODES + 255) / 256, 256, 0, stream>>>(batch, cnt, NNODES);
  pool_scatter<<<(NNODES * HC2 + 255) / 256, 256, 0, stream>>>(hb, batch, gap, gmp,
                                                               NNODES, HC2);
  pool_finalize<<<(NGRAPH * HC2 + 255) / 256, 256, 0, stream>>>(gap, gmp, cnt, NGRAPH, HC2);

  // sol = relu(solvent_fp @ Wsol + bsol)
  dense_bias<<<(NGRAPH * 128 + 255) / 256, 256, 0, stream>>>(solfp, Wsol, bsol, sol,
                                                             NGRAPH, FSOL, 128, 1);
  // fc1 over concat -> bn -> relu -> fc2
  fc1_concat<<<(NGRAPH * 128 + 255) / 256, 256, 0, stream>>>(gap, (const float*)gmp, sol,
                                                             Wfc1, bfc1, zfc);
  bn_small_relu<<<1, 128, 0, stream>>>(zfc, gf, bf, NGRAPH, 128);
  dense_bias<<<1, 128, 0, stream>>>(zfc, Wfc2, bfc2, (float*)d_out, NGRAPH, 128, 2, 0);
}
